// DiffusionNCA_fft2_32847909880200
// MI455X (gfx1250) — matrix operand order repaired and checked by reading, not compiled, not run on hardware
//
#include <hip/hip_runtime.h>

typedef __attribute__((ext_vector_type(16))) _Float16 v16h;
typedef __attribute__((ext_vector_type(8)))  _Float16 v8h;
typedef __attribute__((ext_vector_type(8)))  float    v8f;
typedef __attribute__((ext_vector_type(4)))  unsigned int u32x4;
typedef __attribute__((ext_vector_type(8)))  int      i32x8;
typedef __attribute__((ext_vector_type(4)))  int      i32x4;

#define CH   32
#define Cc   35
#define HID  128
#define Bn   8
#define Hn   256
#define Wn   256
#define HWn  65536
#define NPIX (Bn * HWn)
#define KP   128   // padded K (3*35=105 -> 128) for GEMM1
#define K3C  105

#if defined(__has_builtin)
#if __has_builtin(__builtin_amdgcn_tensor_load_to_lds) && __has_builtin(__builtin_amdgcn_s_wait_tensorcnt)
#define USE_TDM 1
#endif
#endif
#ifndef USE_TDM
#define USE_TDM 0
#endif

// ---------------------------------------------------------------------------
// GroupNorm statistics: one block per sample; extras (pos_x,pos_y,alive) added
// analytically.
// ---------------------------------------------------------------------------
__global__ void k_stats(const float* __restrict__ x, const float* __restrict__ t,
                        float* __restrict__ stats) {
    int b = blockIdx.x;
    int tid = threadIdx.x;
    const float4* xb = (const float4*)(x + (size_t)b * CH * HWn);
    float s = 0.f, ss = 0.f;
    for (int i = tid; i < CH * HWn / 4; i += 256) {
        float4 v = xb[i];
        s  += v.x + v.y + v.z + v.w;
        ss += v.x * v.x + v.y * v.y + v.z * v.z + v.w * v.w;
    }
    __shared__ float rs[256], rss[256];
    rs[tid] = s; rss[tid] = ss;
    __syncthreads();
    for (int o = 128; o > 0; o >>= 1) {
        if (tid < o) { rs[tid] += rs[tid + o]; rss[tid] += rss[tid + o]; }
        __syncthreads();
    }
    if (tid == 0) {
        float lin = 0.f, lin2 = 0.f;
        for (int i = 0; i < Wn; ++i) {
            float v = 1.f - (float)i / 255.f;
            lin += v; lin2 += v * v;
        }
        float tv  = t[b];
        float sum = rs[0]  + 2.f * Hn * lin  + tv * (float)HWn;
        float sq  = rss[0] + 2.f * Hn * lin2 + tv * tv * (float)HWn;
        float tot = (float)Cc * (float)HWn;
        float mean = sum / tot;
        float var  = sq / tot - mean * mean;
        stats[b * 2]     = mean;
        stats[b * 2 + 1] = rsqrtf(var + 1e-5f);
    }
}

// ---------------------------------------------------------------------------
// Normalize: write dx (f32, NCHW, conv input) and activation rows (f16,
// [pixel,128]: k=0..34 normalized channels, k=105..127 zero pad).
// pixel p = b*65536 + h*256 + w
// ---------------------------------------------------------------------------
__global__ void k_norm(const float* __restrict__ x, const float* __restrict__ t,
                       const float* __restrict__ stats,
                       const float* __restrict__ gamma, const float* __restrict__ beta,
                       float* __restrict__ dxf, _Float16* __restrict__ act) {
    int tid = blockIdx.x * 256 + threadIdx.x;      // == pixel p
    int b  = tid >> 16;
    int hw = tid & 65535;
    int h = hw >> 8, w = hw & 255;
    float mean = stats[b * 2], rstd = stats[b * 2 + 1];
    float tv = t[b];
    _Float16* arow = act + (size_t)tid * KP;
    float* drow = dxf + (size_t)b * Cc * HWn + hw;
    for (int c = 0; c < Cc; ++c) {
        float v;
        if (c < CH)           v = x[(((size_t)b * CH + c) << 16) + hw];
        else if (c == CH)     v = 1.f - (float)w / 255.f;   // pos_x
        else if (c == CH + 1) v = 1.f - (float)h / 255.f;   // pos_y
        else                  v = tv;                        // alive_rate
        float xn = (v - mean) * rstd * gamma[c] + beta[c];
        drow[(size_t)c * HWn] = xn;
        arow[c] = (_Float16)xn;
    }
    for (int k = K3C; k < KP; ++k) arow[k] = (_Float16)0.f;
}

// ---------------------------------------------------------------------------
// Depthwise 7x7 conv, reflect padding, two filter banks; writes f16 into
// activation k=35..69 (conv0) and k=70..104 (conv1).
// ---------------------------------------------------------------------------
__global__ void k_conv(const float* __restrict__ dxf,
                       const float* __restrict__ cw0, const float* __restrict__ cb0,
                       const float* __restrict__ cw1, const float* __restrict__ cb1,
                       _Float16* __restrict__ act) {
    __shared__ float tile[22][22];
    __shared__ float wk[2][49];
    int bc  = blockIdx.z;        // b*35 + c
    int c   = bc % Cc;
    int tid = threadIdx.x;
    int tx = tid & 15, ty = tid >> 4;
    int hb = blockIdx.y * 16, wb = blockIdx.x * 16;
    const float* src = dxf + (size_t)bc * HWn;
    if (tid < 98) {
        wk[tid / 49][tid % 49] = (tid < 49 ? cw0 : cw1)[c * 49 + (tid % 49)];
    }
    for (int i = tid; i < 22 * 22; i += 256) {
        int th = i / 22, tw = i % 22;
        int gh = hb + th - 3, gw = wb + tw - 3;
        gh = gh < 0 ? -gh : (gh > 255 ? 510 - gh : gh);
        gw = gw < 0 ? -gw : (gw > 255 ? 510 - gw : gw);
        tile[th][tw] = src[gh * Wn + gw];
    }
    __syncthreads();
    float a0 = 0.f, a1 = 0.f;
    #pragma unroll
    for (int i = 0; i < 7; ++i)
        #pragma unroll
        for (int j = 0; j < 7; ++j) {
            float v = tile[ty + i][tx + j];
            a0 += v * wk[0][i * 7 + j];
            a1 += v * wk[1][i * 7 + j];
        }
    a0 += cb0[c]; a1 += cb1[c];
    int b = bc / Cc;
    size_t p = (size_t)b * HWn + (hb + ty) * Wn + (wb + tx);
    act[p * KP + Cc + c]     = (_Float16)a0;
    act[p * KP + 2 * Cc + c] = (_Float16)a1;
}

// ---------------------------------------------------------------------------
// Fused per-pixel MLP via WMMA. 8 waves/block, 16 pixels/wave (128 px/block).
// The block's contiguous 32 KB activation slab (128 px x 256 B) is DMA'd into
// LDS by the Tensor Data Mover (issued by wave 0, overlapped with the weight
// pack), then A-fragments come from LDS.
// GEMM1: 4 K-tiles x 8 N-tiles of wmma_f32_16x16x32_f16; hidden staged through
// LDS (f16) to re-lay C/D layout -> A layout; GEMM2: 4 x 2 WMMAs; mask+residual.
// ---------------------------------------------------------------------------
__global__ void k_gemm(const _Float16* __restrict__ act,
                       const float* __restrict__ fc0w, const float* __restrict__ fc0b,
                       const float* __restrict__ fc1w,
                       const float* __restrict__ rmask,   // [B,W,H] for this step
                       const float* __restrict__ xin, float* __restrict__ xout) {
    __shared__ __attribute__((aligned(32))) _Float16 w0B[4 * 8 * 32 * 16]; // 32 KB
    __shared__ __attribute__((aligned(32))) _Float16 w1B[4 * 2 * 32 * 16]; //  8 KB
    __shared__ __attribute__((aligned(32))) _Float16 hid[8 * 16 * HID];    // 32 KB
    __shared__ __attribute__((aligned(32))) _Float16 ldsA[128 * KP];       // 32 KB
    int tid = threadIdx.x;

#if USE_TDM
    if (tid < 32) {
        // Tensor DMA: contiguous 16384 x 2B tile  act[block*128*KP ..) -> ldsA.
        unsigned long long gaddr =
            (unsigned long long)(uintptr_t)(act + (size_t)blockIdx.x * 128 * KP);
        unsigned ldsoff = (unsigned)(uintptr_t)(void*)&ldsA[0];  // LDS byte offset
        // D# group0: count=1, lds_addr[63:32], global_addr[120:64], type=2 [127:126]
        u32x4 g0 = { 1u, ldsoff,
                     (unsigned)(gaddr & 0xffffffffull),
                     (unsigned)((gaddr >> 32) & 0x01ffffffull) | 0x80000000u };
        // D# group1: data_size=2B (code 1) @ [17:16]; tensor_dim0=16384 @ [79:48];
        //            tile_dim0=16384 @ [127:112]; tensor_dim0_stride=16384 @ [207:160]
        i32x8 g1 = { (int)(1u << 16), (int)(16384u << 16), 0, (int)(16384u << 16),
                     0, 16384, 0, 0 };
        i32x4 g2 = { 0, 0, 0, 0 };
        i32x4 g3 = { 0, 0, 0, 0 };
#if defined(__clang_major__) && (__clang_major__ >= 23)
        i32x8 g4 = { 0, 0, 0, 0, 0, 0, 0, 0 };
        __builtin_amdgcn_tensor_load_to_lds(g0, g1, g2, g3, g4, 0);
#else
        __builtin_amdgcn_tensor_load_to_lds(g0, g1, g2, g3, 0);
#endif
    }
#endif

    // Pack fc0_w [128,105] into B-matrix layout: B[K,N] = W0[N,K], K padded to 128.
    // B layout (16-bit, 32x16): lane = N + 16*h, half j: K = 8h + j (j<8), 16 + 8h + (j-8) (j>=8).
    for (int idx = tid; idx < 4 * 8 * 32 * 16; idx += 256) {
        int j    = idx & 15;
        int lane = (idx >> 4) & 31;
        int nt   = (idx >> 9) & 7;
        int kt   = idx >> 12;
        int Kg = kt * 32 + ((j >> 3) << 4) + ((lane >> 4) << 3) + (j & 7);
        int Ng = nt * 16 + (lane & 15);
        w0B[idx] = (Kg < K3C) ? (_Float16)fc0w[Ng * K3C + Kg] : (_Float16)0.f;
    }
    // Pack fc1_w [32,128] the same way.
    for (int idx = tid; idx < 4 * 2 * 32 * 16; idx += 256) {
        int j    = idx & 15;
        int lane = (idx >> 4) & 31;
        int nt   = (idx >> 9) & 1;
        int kt   = idx >> 10;
        int Kg = kt * 32 + ((j >> 3) << 4) + ((lane >> 4) << 3) + (j & 7);
        int Ng = nt * 16 + (lane & 15);
        w1B[idx] = (_Float16)fc1w[Ng * HID + Kg];
    }

#if USE_TDM
    if (tid < 32) __builtin_amdgcn_s_wait_tensorcnt(0);
#else
    {   // Fallback: cooperative copy of the activation slab into LDS.
        const v8h* src = (const v8h*)(act + (size_t)blockIdx.x * 128 * KP);
        v8h* dst = (v8h*)&ldsA[0];
        for (int i = tid; i < 128 * KP / 8; i += 256) dst[i] = src[i];
    }
#endif
    __syncthreads();

    int wave = tid >> 5;
    int lane = tid & 31;
    size_t p0 = (size_t)blockIdx.x * 128 + (size_t)wave * 16;
    int off = (lane >= 16) ? 8 : 0;   // A layout: hi half-wave owns K+8 / K+24
    int M   = lane & 15;

    v8f acc[8] = {};
    union AV { v16h v; v8h h[2]; } a;
    const _Float16* arow = &ldsA[(wave * 16 + M) * KP];
    __builtin_prefetch(act + ((size_t)blockIdx.x + 1) * 128 * KP, 0, 0); // next slab

    #pragma unroll
    for (int kt = 0; kt < 4; ++kt) {
        a.h[0] = *(const v8h*)(arow + kt * 32 + off);        // K = kt*32+off .. +7
        a.h[1] = *(const v8h*)(arow + kt * 32 + 16 + off);   // K = kt*32+16+off .. +7
        #pragma unroll
        for (int nt = 0; nt < 8; ++nt) {
            v16h bfrag = *(const v16h*)&w0B[((kt * 8 + nt) * 32 + lane) * 16];
            acc[nt] = __builtin_amdgcn_wmma_f32_16x16x32_f16(
                false, a.v, false, bfrag, (short)0, acc[nt], false, false);
        }
    }

    // Epilogue 1: bias + leaky_relu(0.01), spill hidden to LDS as f16.
    // C/D layout: lane l, slot r -> M = r + 8*(l/16), N = l%16.
    int Mbase = (lane >> 4) << 3;
    int Ncol  = lane & 15;
    _Float16* hrow = hid + wave * 16 * HID;
    #pragma unroll
    for (int nt = 0; nt < 8; ++nt) {
        float bia = fc0b[nt * 16 + Ncol];
        #pragma unroll
        for (int r = 0; r < 8; ++r) {
            float v = acc[nt][r] + bia;
            v = v > 0.f ? v : 0.01f * v;
            hrow[(Mbase + r) * HID + nt * 16 + Ncol] = (_Float16)v;
        }
    }
    __syncthreads();

    v8f acc2[2] = {};
    #pragma unroll
    for (int kt = 0; kt < 4; ++kt) {
        union AV a2;
        a2.h[0] = *(const v8h*)(hrow + M * HID + kt * 32 + off);
        a2.h[1] = *(const v8h*)(hrow + M * HID + kt * 32 + 16 + off);
        #pragma unroll
        for (int nt = 0; nt < 2; ++nt) {
            v16h bfrag = *(const v16h*)&w1B[((kt * 2 + nt) * 32 + lane) * 16];
            acc2[nt] = __builtin_amdgcn_wmma_f32_16x16x32_f16(
                false, a2.v, false, bfrag, (short)0, acc2[nt], false, false);
        }
    }

    // Epilogue 2: fire mask + residual add into x.
    #pragma unroll
    for (int nt = 0; nt < 2; ++nt) {
        int ch = nt * 16 + Ncol;
        #pragma unroll
        for (int r = 0; r < 8; ++r) {
            size_t p = p0 + (size_t)(Mbase + r);
            int b  = (int)(p >> 16);
            int hw = (int)(p & 65535);
            int h = hw >> 8, w = hw & 255;
            float rm = rmask[((size_t)b << 16) + ((size_t)w << 8) + h];
            float m  = rm > 0.5f ? 1.f : 0.f;
            size_t xi = (((size_t)b * CH + ch) << 16) + hw;
            xout[xi] = xin[xi] + acc2[nt][r] * m;
        }
    }
}

// ---------------------------------------------------------------------------
extern "C" void kernel_launch(void* const* d_in, const int* in_sizes, int n_in,
                              void* d_out, int out_size, void* d_ws, size_t ws_size,
                              hipStream_t stream) {
    const float* x     = (const float*)d_in[0];
    const float* t     = (const float*)d_in[1];
    const float* rmask = (const float*)d_in[2];
    // d_in[3] = steps (==2, compile-time constant here)
    const float* gamma = (const float*)d_in[4];
    const float* beta  = (const float*)d_in[5];
    const float* c0w   = (const float*)d_in[6];
    const float* c0b   = (const float*)d_in[7];
    const float* c1w   = (const float*)d_in[8];
    const float* c1b   = (const float*)d_in[9];
    const float* f0w   = (const float*)d_in[10];
    const float* f0b   = (const float*)d_in[11];
    const float* f1w   = (const float*)d_in[12];
    float* out = (float*)d_out;

    char* ws = (char*)d_ws;
    float*    stats = (float*)ws;                                   // 64 B
    float*    dxf   = (float*)(ws + 256);                           // 8*35*65536*4 = 73.4 MB
    _Float16* act   = (_Float16*)(ws + 256 + (size_t)Bn * Cc * HWn * 4); // 524288*128*2 = 134 MB

    for (int s = 0; s < 2; ++s) {
        const float* xin = (s == 0) ? x : (const float*)out;
        k_stats<<<Bn, 256, 0, stream>>>(xin, t, stats);
        k_norm <<<NPIX / 256, 256, 0, stream>>>(xin, t, stats, gamma, beta, dxf, act);
        k_conv <<<dim3(Wn / 16, Hn / 16, Bn * Cc), 256, 0, stream>>>(dxf, c0w, c0b, c1w, c1b, act);
        k_gemm <<<NPIX / 128, 256, 0, stream>>>(act, f0w, f0b, f1w,
                                                rmask + (size_t)s * Bn * HWn, xin, out);
    }
}
